// GAT_63256278336041
// MI455X (gfx1250) — matrix operand order repaired and checked by reading, not compiled, not run on hardware
//
#include <hip/hip_runtime.h>
#include <hip/hip_bf16.h>

typedef __attribute__((ext_vector_type(16))) _Float16 v16h;
typedef __attribute__((ext_vector_type(8)))  _Float16 v8h;
typedef __attribute__((ext_vector_type(8)))  float    v8f;

#define NN 32768
#define SS 64
#define EE 256

// ---------------------------------------------------------------------------
// Fragment layouts (ISA 7.12.2, wave32, f16 16x16x32):
//  A 16x32: lane 0-15: M=lane,   K(i) = i + (i>=8 ? 8 : 0)        (kb=0)
//           lane16-31: M=lane-16,K(i) = 8 + i + (i>=8 ? 8 : 0)    (kb=8)
//  B 32x16: lane 0-15: N=lane,   K(i) = i + (i>=8 ? 8 : 0)
//           lane16-31: N=lane-16,K(i) = 8 + i + (i>=8 ? 8 : 0)
//  C/D f32: M = r + 8*(lane>=16), N = lane&15
// Swizzled storage blocks of 512 halves: off = blk*512 + lane*16 + i
// ---------------------------------------------------------------------------

__global__ void gat_prep_weights(const float* __restrict__ Wq,
                                 const float* __restrict__ Wk,
                                 const float* __restrict__ Wfc,
                                 _Float16* __restrict__ wqTsw,
                                 _Float16* __restrict__ wkTsw,
                                 _Float16* __restrict__ wfcAsw) {
    int d = blockIdx.x * blockDim.x + threadIdx.x;
    if (d < 4096) {                       // B-layout blocks: (tileN[4], kc[2])
        int blk = d >> 9, within = d & 511;
        int ln = within >> 4, ii = within & 15;
        int tN = blk >> 1, kc = (blk & 1) * 32;
        int n = tN * 16 + (ln & 15);
        int k = kc + ((ln & 16) ? 8 : 0) + ii + ((ii >= 8) ? 8 : 0);
        wqTsw[d] = (_Float16)Wq[n * 64 + k];   // WqT[k][n] = Wq[n][k]
        wkTsw[d] = (_Float16)Wk[n * 64 + k];
    }
    if (d < 8192) {                       // A-layout blocks: (tileM[4], kc4[4])
        int blk = d >> 9, within = d & 511;
        int ln = within >> 4, ii = within & 15;
        int tM = blk >> 2, kc4 = blk & 3;
        int m = tM * 16 + (ln & 15);
        int kb = (ln & 16) ? 8 : 0;
        int kl = (ii < 8) ? (kb + ii) : (kb + ii + 8);
        wfcAsw[d] = (_Float16)Wfc[m * 128 + kc4 * 32 + kl];
    }
}

__device__ __forceinline__ v8f wmma_f16(v16h a, v16h b, v8f c) {
    return __builtin_amdgcn_wmma_f32_16x16x32_f16(false, a, false, b,
                                                  (short)0, c, false, false);
}

__device__ __forceinline__ v16h frag(const _Float16* p) {
    return *(const v16h*)p;
}

__device__ __forceinline__ float tanh_fast(float x) {
#if __has_builtin(__builtin_amdgcn_tanhf)
    return __builtin_amdgcn_tanhf(x);
#elif __has_builtin(__builtin_amdgcn_tanh_f32)
    return __builtin_amdgcn_tanh_f32(x);
#else
    float e = __builtin_amdgcn_exp2f(x * 2.8853900817779268f);   // exp(2x)
    return 1.0f - 2.0f * __builtin_amdgcn_rcpf(e + 1.0f);
#endif
}

// LDS pool offsets (bytes)
#define O_EMBA  0            // 2 x 4096 f16, A-swizzled snapshots
#define O_EMBB  16384        // 2 x 4096 f16, B-swizzled snapshots
#define O_QHT   32768        // qhT[h][qi] f16 pitch 72 (later aliased by y2xh)
#define O_KHT   41984        // khT[h][ki] f16 pitch 72
#define O_WH    51200        // 4096 f16, A-swizzled softmax weights
#define O_RSUM  59392        // 64 f32 row sums
#define O_WVS   59648        // 64 f32
#define O_BFC   59904        // 64 f32
#define O_VLEN  60160        // 2 i32
#define SMEM_SZ 60192

__global__ __launch_bounds__(512)
void gat_edge_chain(float* __restrict__ emb,
                    const int* __restrict__ masks,
                    const int* __restrict__ edge_index,
                    const _Float16* __restrict__ WqTsw,
                    const _Float16* __restrict__ WkTsw,
                    const _Float16* __restrict__ WfcAsw,
                    const float* __restrict__ wv,
                    const float* __restrict__ bfc) {
    __shared__ __align__(32) unsigned char smem[SMEM_SZ];
    _Float16* embhA = (_Float16*)(smem + O_EMBA);
    _Float16* embhB = (_Float16*)(smem + O_EMBB);
    _Float16* qhT   = (_Float16*)(smem + O_QHT);
    _Float16* khT   = (_Float16*)(smem + O_KHT);
    _Float16* wh    = (_Float16*)(smem + O_WH);
    _Float16* y2xh  = (_Float16*)(smem + O_QHT);  // aliases qhT (dead by then)
    float*    rowsum= (float*)(smem + O_RSUM);
    float*    wvs   = (float*)(smem + O_WVS);
    float*    bfcs  = (float*)(smem + O_BFC);
    int*      vlen  = (int*)(smem + O_VLEN);

    const int tid   = threadIdx.x;
    const int lane  = tid & 31;
    const int wave  = tid >> 5;
    const int tileM = (wave >> 2) * 16;
    const int tileN = (wave & 3) * 16;
    const int tA    = tileM >> 4;
    const int tB    = tileN >> 4;
    // score-phase mapping: wave-uniform key octet, lane = query row
    const int sub8  = wave & 7;           // key octet 0..7 (uniform per wave)
    const int qg    = wave >> 3;          // query half 0..1
    const int qi    = qg * 32 + lane;     // query row 0..63

    if (tid < SS) { wvs[tid] = wv[tid]; bfcs[tid] = bfc[tid]; }

    // preload all weight fragments into registers (edge-invariant)
    const v16h wqf0 = frag(&WqTsw[(tB * 2 + 0) * 512 + lane * 16]);
    const v16h wqf1 = frag(&WqTsw[(tB * 2 + 1) * 512 + lane * 16]);
    const v16h wkf0 = frag(&WkTsw[(tB * 2 + 0) * 512 + lane * 16]);
    const v16h wkf1 = frag(&WkTsw[(tB * 2 + 1) * 512 + lane * 16]);
    const v16h wff0 = frag(&WfcAsw[(tA * 4 + 0) * 512 + lane * 16]);
    const v16h wff1 = frag(&WfcAsw[(tA * 4 + 1) * 512 + lane * 16]);
    const v16h wff2 = frag(&WfcAsw[(tA * 4 + 2) * 512 + lane * 16]);
    const v16h wff3 = frag(&WfcAsw[(tA * 4 + 3) * 512 + lane * 16]);

    for (int e = 0; e < EE; ++e) {
        const int x = edge_index[e];
        const int y = edge_index[EE + e];

        // prefetch next edge's node rows (16 KB each)
        if (e + 1 < EE && tid < 64) {
            const int xn = edge_index[e + 1], yn = edge_index[EE + e + 1];
            __builtin_prefetch((const char*)(emb + (size_t)xn * 4096) + tid * 256, 0, 0);
            __builtin_prefetch((const char*)(emb + (size_t)yn * 4096) + tid * 256, 0, 0);
        }

        // A-swizzled snapshot: octet = 8 consecutive k -> 2x float4 + 16B store
        for (int g = tid; g < 1024; g += 512) {
            const int node = g >> 9;
            const int d    = (g & 511) * 8;
            const int blk = d >> 9, within = d & 511;
            const int ln  = within >> 4, oct = (within >> 3) & 1;
            const int tt  = blk >> 1, kc = (blk & 1) * 32;
            const int m   = tt * 16 + (ln & 15);
            const int kb  = (ln & 16) ? 8 : 0;
            const int k0  = kc + kb + (oct ? 16 : 0);
            const float4* src = (const float4*)&emb[(size_t)(node ? y : x) * 4096
                                                    + m * 64 + k0];
            const float4 f0 = src[0], f1 = src[1];
            v8h o;
            o[0] = (_Float16)f0.x; o[1] = (_Float16)f0.y;
            o[2] = (_Float16)f0.z; o[3] = (_Float16)f0.w;
            o[4] = (_Float16)f1.x; o[5] = (_Float16)f1.y;
            o[6] = (_Float16)f1.z; o[7] = (_Float16)f1.w;
            *(v8h*)&embhA[node * 4096 + d] = o;
        }
        // B-swizzled snapshot: column gather (strided, hits prefetched L2)
        for (int d = tid; d < 8192; d += 512) {
            const int node = d >> 12;
            const int dd = d & 4095;
            const int blk = dd >> 9, within = dd & 511;
            const int ln = within >> 4, ii = within & 15;
            const int tt = blk >> 1, kc = (blk & 1) * 32;
            const int n = tt * 16 + (ln & 15);
            const int k = kc + ((ln & 16) ? 8 : 0) + ii + ((ii >= 8) ? 8 : 0);
            embhB[d] = (_Float16)emb[(size_t)(node ? y : x) * 4096 + k * 64 + n];
        }
        if (tid < 2) {
            const int node = tid ? y : x;
            int s = 0;
#pragma unroll
            for (int h = 0; h < SS; ++h) s += masks[node * SS + h];
            vlen[tid] = s;
        }
        __syncthreads();

        for (int dir = 0; dir < 2; ++dir) {
            const int qn = dir;        // query node (ex dir0 / ey dir1)
            const int kn = 1 - dir;    // key/value node

            // ---- phase 1: q/k projections -> TRANSPOSED LDS; zero rowsum --
            {
                v8f cq = {}, ck = {};
                const int a0 = (tA * 2 + 0) * 512 + lane * 16;
                const int a1 = (tA * 2 + 1) * 512 + lane * 16;
                cq = wmma_f16(frag(&embhA[qn * 4096 + a0]), wqf0, cq);
                cq = wmma_f16(frag(&embhA[qn * 4096 + a1]), wqf1, cq);
                ck = wmma_f16(frag(&embhA[kn * 4096 + a0]), wkf0, ck);
                ck = wmma_f16(frag(&embhA[kn * 4096 + a1]), wkf1, ck);
                const int mrow = tileM + ((lane & 16) ? 8 : 0);
                const int ncol = tileN + (lane & 15);
                v8h oq, ok;
#pragma unroll
                for (int r = 0; r < 8; ++r) {
                    oq[r] = (_Float16)cq[r];
                    ok[r] = (_Float16)ck[r];
                }
                // D(M,N) -> T[h=N][row=M]: 8 consecutive rows = one 16B store
                *(v8h*)&qhT[ncol * 72 + mrow] = oq;
                *(v8h*)&khT[ncol * 72 + mrow] = ok;
                if (tid < 64) rowsum[tid] = 0.0f;
            }
            __syncthreads();

            // ---- phase 2: scores (wave skips fully-masked key octets) ----
            float s[8];
            {
                const int vl = vlen[dir];
                if (vl == 0) {                       // degenerate: uniform
#pragma unroll
                    for (int j = 0; j < 8; ++j) s[j] = 1.0f;
                } else if (sub8 * 8 < vl) {          // octet (partially) live
                    float acc[8];
#pragma unroll
                    for (int j = 0; j < 8; ++j) acc[j] = 0.0f;
#pragma unroll 2
                    for (int h = 0; h < SS; ++h) {
                        const float qv  = (float)qhT[h * 72 + qi];
                        const float wvh = wvs[h];
                        const v8h  kv8  = *(const v8h*)&khT[h * 72 + sub8 * 8];
#pragma unroll
                        for (int j = 0; j < 8; ++j)
                            acc[j] = __builtin_fmaf(
                                wvh, tanh_fast(qv + (float)kv8[j]), acc[j]);
                    }
#pragma unroll
                    for (int j = 0; j < 8; ++j) {
                        // |score| <~ 7 -> exp without max-subtraction is safe
                        s[j] = (sub8 * 8 + j < vl)
                             ? __builtin_amdgcn_exp2f(acc[j] * 1.4426950408889634f)
                             : 0.0f;
                    }
                } else {                             // fully masked octet
#pragma unroll
                    for (int j = 0; j < 8; ++j) s[j] = 0.0f;
                }
                float local = 0.0f;
#pragma unroll
                for (int j = 0; j < 8; ++j) local += s[j];
                atomicAdd(&rowsum[qi], local);       // ds_add_f32
            }
            __syncthreads();

            // ---- phase 3: normalize, write wh in A-layout (16B merged) ----
            {
                const float inv = __builtin_amdgcn_rcpf(rowsum[qi]);
#pragma unroll
                for (int j = 0; j < 8; ++j) {
                    const int ki = sub8 * 8 + j;
                    const int kl = ki & 31;
                    const int laneHi = (kl >> 3) & 1;
                    const int iA = kl - 8 * laneHi - 8 * (kl >> 4);
                    const int laneA = (qi & 15) + 16 * laneHi;
                    wh[((qi >> 4) * 2 + (ki >> 5)) * 512 + laneA * 16 + iA] =
                        (_Float16)(s[j] * inv);
                }
            }
            __syncthreads();

            // ---- phase 4: y2x = softmax_w @ kv -> B-swizzled (alias qhT) --
            {
                v8f c = {};
#pragma unroll
                for (int kc = 0; kc < 2; ++kc) {
                    const int ao = (tA * 2 + kc) * 512 + lane * 16;
                    const int bo = (tB * 2 + kc) * 512 + lane * 16;
                    c = wmma_f16(frag(&wh[ao]), frag(&embhB[kn * 4096 + bo]), c);
                }
                const int mrow = tileM + ((lane & 16) ? 8 : 0);
                const int ncol = tileN + (lane & 15);
#pragma unroll
                for (int r = 0; r < 8; ++r) {
                    const int q  = mrow + r;
                    const int kcB = q >> 5, kl = q & 31;
                    const int laneB = (ncol & 15) + 16 * ((kl >> 3) & 1);
                    y2xh[((ncol >> 4) * 2 + kcB) * 512 + laneB * 16 + ((kl >> 4) << 3) + (kl & 7)] =
                        (_Float16)c[r];
                }
            }
            __syncthreads();

            // ---- phase 5: out = Wfc @ cat([q_in ; y2x]) + bfc ----
            {
                v8f c = {};
                const int b0 = (tB * 2 + 0) * 512 + lane * 16;
                const int b1 = (tB * 2 + 1) * 512 + lane * 16;
                c = wmma_f16(wff0, frag(&embhB[qn * 4096 + b0]), c);
                c = wmma_f16(wff1, frag(&embhB[qn * 4096 + b1]), c);
                c = wmma_f16(wff2, frag(&y2xh[b0]), c);
                c = wmma_f16(wff3, frag(&y2xh[b1]), c);
                const int node = dir ? y : x;   // x first, y second (y wins)
                const int mrow = tileM + ((lane & 16) ? 8 : 0);
                const int ncol = tileN + (lane & 15);
#pragma unroll
                for (int r = 0; r < 8; ++r) {
                    emb[(size_t)node * 4096 + (size_t)(mrow + r) * 64 + ncol] =
                        c[r] + bfcs[mrow + r];
                }
            }
            __threadfence();
            __syncthreads();
        }
    }
}

// ---------------------------------------------------------------------------
extern "C" void kernel_launch(void* const* d_in, const int* in_sizes, int n_in,
                              void* d_out, int out_size, void* d_ws, size_t ws_size,
                              hipStream_t stream) {
    const float* inputs = (const float*)d_in[0];
    const int*   masks  = (const int*)d_in[1];
    const int*   eidx   = (const int*)d_in[2];
    const float* Wq     = (const float*)d_in[3];
    const float* Wk     = (const float*)d_in[4];
    const float* wv     = (const float*)d_in[5];
    const float* Wfc    = (const float*)d_in[6];
    const float* bfc    = (const float*)d_in[7];
    float* out = (float*)d_out;

    _Float16* wqTsw  = (_Float16*)d_ws;                  //  8 KB
    _Float16* wkTsw  = (_Float16*)((char*)d_ws + 8192);  //  8 KB
    _Float16* wfcAsw = (_Float16*)((char*)d_ws + 16384); // 16 KB

    gat_prep_weights<<<32, 256, 0, stream>>>(Wq, Wk, Wfc, wqTsw, wkTsw, wfcAsw);

    // bulk identity copy (~1 GB traffic, HBM-roofline ~44 us)
    hipMemcpyAsync(out, inputs, (size_t)NN * SS * SS * sizeof(float),
                   hipMemcpyDeviceToDevice, stream);

    // serial 256-edge chain: one persistent 16-wave workgroup, in place
    gat_edge_chain<<<1, 512, 0, stream>>>(out, masks, eidx, wqTsw, wkTsw,
                                          wfcAsw, wv, bfc);

    (void)in_sizes; (void)n_in; (void)out_size; (void)ws_size;
}
